// SeparableFiberBundleConvNext_36541581754379
// MI455X (gfx1250) — matrix operand order repaired and checked by reading, not compiled, not run on hardware
//
#include <hip/hip_runtime.h>

#define NN   10000
#define OO   16
#define CC   64
#define EE   100000
#define KDD  64
#define HID  256
#define EPW  4                 // edges per wave in k_edge
#define LN_EPS 1e-6f
#define LS_VAL 1e-6f

typedef __attribute__((ext_vector_type(16))) __bf16 v16bf;
typedef __attribute__((ext_vector_type(8)))  float  v8f;

static __device__ __forceinline__ unsigned short to_bf16(float f) {
    union { float f; unsigned u; } v; v.f = f;
    unsigned r = v.u + 0x7FFFu + ((v.u >> 16) & 1u);   // round-to-nearest-even
    return (unsigned short)(r >> 16);
}

// A-matrix (16-bit, 16x32) LDS slot for value at (row m, k-within-tile r):
// reader lane l holds m=l&15, kbase=8*(l>=16), elem i -> K = kbase + 16*(i>>3) + (i&7)
static __device__ __forceinline__ int a_slot(int m, int r) {
    int lane = m + (((r >> 3) & 1) << 4);
    int i    = (r & 7) + ((r >> 4) << 3);
    return lane * 16 + i;
}

static __device__ __forceinline__ v8f v8f_zero() {
    v8f z = {0.f,0.f,0.f,0.f,0.f,0.f,0.f,0.f};
    return z;
}

// ---------------- prep kernels ----------------

__global__ void k_zero(float4* __restrict__ p) {
    p[(size_t)blockIdx.x * blockDim.x + threadIdx.x] = make_float4(0.f,0.f,0.f,0.f);
}

// fk[p,o,c] = sum_k fiber_basis[p,o,k] * fiber_w[k,c]
__global__ void k_fk(const float* __restrict__ fb, const float* __restrict__ fw,
                     float* __restrict__ fk) {
    int t  = blockIdx.x * blockDim.x + threadIdx.x;   // 16384 total
    int c  = t & 63;
    int po = t >> 6;
    float s = 0.f;
    #pragma unroll 8
    for (int k = 0; k < KDD; ++k) s += fb[po * KDD + k] * fw[k * CC + c];
    fk[t] = s;
}

// Pre-swizzle a row-major f32 [K][Nn] weight into bf16 B-matrix fragments.
// Fragment f = (k/32)*(Nn/16) + (n/16); lane = n%16 + 16*((k%16)/8); elem = k%8 + 8*((k%32)/16).
__global__ void k_swz(const float* __restrict__ src, unsigned short* __restrict__ dst,
                      int K, int Nn) {
    int t = blockIdx.x * blockDim.x + threadIdx.x;
    if (t >= K * Nn) return;
    int k = t / Nn, n = t % Nn;
    int f    = (k >> 5) * (Nn >> 4) + (n >> 4);
    int lane = (n & 15) + (((k >> 3) & 1) << 4);
    int i    = (k & 7) + (((k >> 4) & 1) << 3);
    dst[f * 512 + lane * 16 + i] = to_bf16(src[t]);
}

// ---------------- edge conv: per-edge 16x64x64 GEMM + gather/modulate/scatter ----------------

__global__ void k_edge(const float* __restrict__ x,
                       const float* __restrict__ basis,
                       const int*   __restrict__ ei,
                       const unsigned short* __restrict__ kws,
                       float* __restrict__ x1) {
    __shared__ unsigned short Ast[8][2][512];
    const int wave  = threadIdx.x >> 5;
    const int lane  = threadIdx.x & 31;
    const int ebase = (blockIdx.x * 8 + wave) * EPW;   // grid = E/(8*EPW) exactly

    // kernel_w B-fragments live in registers for all EPW edges
    v16bf bf[2][4];
    #pragma unroll
    for (int kt2 = 0; kt2 < 2; ++kt2)
        #pragma unroll
        for (int nt = 0; nt < 4; ++nt)
            bf[kt2][nt] = *(const v16bf*)(kws + ((kt2 * 4 + nt) * 512 + lane * 16));

    const int o  = lane >> 1;
    const int kt = lane & 1;
    const int nl = lane & 15, hi = lane >> 4;

    for (int it = 0; it < EPW; ++it) {
        const int e = ebase + it;
        const float* src = basis + (size_t)e * (OO * KDD) + o * KDD + (kt << 5);

        // prefetch next edge's basis tile while this one computes
        if (it + 1 < EPW)
            __builtin_prefetch((const void*)(src + OO * KDD), 0, 1);

        // gather the 16x64 basis tile (this lane: 32 contiguous floats)
        float4 vv[8];
        #pragma unroll
        for (int j = 0; j < 8; ++j) vv[j] = *(const float4*)(src + j * 4);
        #pragma unroll
        for (int j = 0; j < 8; ++j) {
            Ast[wave][kt][a_slot(o, j * 4 + 0)] = to_bf16(vv[j].x);
            Ast[wave][kt][a_slot(o, j * 4 + 1)] = to_bf16(vv[j].y);
            Ast[wave][kt][a_slot(o, j * 4 + 2)] = to_bf16(vv[j].z);
            Ast[wave][kt][a_slot(o, j * 4 + 3)] = to_bf16(vv[j].w);
        }
        __syncthreads();

        v16bf a0 = *(const v16bf*)&Ast[wave][0][lane * 16];
        v16bf a1 = *(const v16bf*)&Ast[wave][1][lane * 16];

        v8f acc[4];
        #pragma unroll
        for (int nt = 0; nt < 4; ++nt) {
            v8f c = v8f_zero();
            c = __builtin_amdgcn_wmma_f32_16x16x32_bf16(false, a0, false, bf[0][nt], (short)0, c, false, false);
            c = __builtin_amdgcn_wmma_f32_16x16x32_bf16(false, a1, false, bf[1][nt], (short)0, c, false, false);
            acc[nt] = c;
        }

        const int srcn = ei[e];
        const int dstn = ei[EE + e];
        const float* xs = x  + (size_t)srcn * (OO * CC);
        float*       xd = x1 + (size_t)dstn * (OO * CC);

        // phase 1: gather all modulation values (independent loads -> one clause)
        float xv[4][8];
        #pragma unroll
        for (int nt = 0; nt < 4; ++nt)
            #pragma unroll
            for (int r = 0; r < 8; ++r)
                xv[nt][r] = xs[(r + 8 * hi) * CC + nt * 16 + nl];

        // phase 2: scatter-add (hardware f32 atomics, L2-resident destination)
        #pragma unroll
        for (int nt = 0; nt < 4; ++nt)
            #pragma unroll
            for (int r = 0; r < 8; ++r)
                unsafeAtomicAdd(&xd[(r + 8 * hi) * CC + nt * 16 + nl],
                                acc[nt][r] * xv[nt][r]);
    }
}

// ---------------- per-node: fiber conv + LN + ConvNeXt MLP + residual ----------------

__global__ void k_node(const float* __restrict__ x,
                       const float* __restrict__ x1,
                       const float* __restrict__ fk,
                       const float* __restrict__ conv_bias,
                       const float* __restrict__ ln_scale,
                       const float* __restrict__ ln_bias,
                       const unsigned short* __restrict__ w1s,
                       const float* __restrict__ b1,
                       const unsigned short* __restrict__ w2s,
                       const float* __restrict__ b2,
                       const float* __restrict__ layer_scale,
                       float* __restrict__ out) {
    __shared__ unsigned short Ast[8][2][512];
    __shared__ unsigned short Gst[8][512];
    const int wave = threadIdx.x >> 5;
    const int lane = threadIdx.x & 31;
    const int node = blockIdx.x * 8 + wave;            // grid = N/8 exactly

    const int p  = lane >> 1;
    const int c0 = (lane & 1) << 5;
    const float* xr = x1 + (size_t)node * (OO * CC);

    // fiber (orientation) conv, depthwise over channels: x2[p,c] = (1/O) sum_o x1[o,c]*fk[p,o,c]
    float h[32];
    #pragma unroll
    for (int j = 0; j < 32; ++j) h[j] = 0.f;
    for (int o = 0; o < OO; ++o) {
        const float* xa = xr + o * CC + c0;
        const float* fa = fk + ((p * OO + o) * CC) + c0;
        #pragma unroll
        for (int j = 0; j < 32; j += 4) {
            float4 xv = *(const float4*)(xa + j);
            float4 fv = *(const float4*)(fa + j);
            h[j + 0] += xv.x * fv.x;
            h[j + 1] += xv.y * fv.y;
            h[j + 2] += xv.z * fv.z;
            h[j + 3] += xv.w * fv.w;
        }
    }
    // bias + LayerNorm stats (this lane owns 32 of the 64 channels of row p)
    float s = 0.f, sq = 0.f;
    #pragma unroll
    for (int j = 0; j < 32; ++j) {
        float v = h[j] * (1.f / OO) + conv_bias[c0 + j];
        h[j] = v; s += v; sq += v * v;
    }
    s  += __shfl_xor(s, 1, 32);
    sq += __shfl_xor(sq, 1, 32);
    const float mu  = s * (1.f / CC);
    const float inv = rsqrtf(sq * (1.f / CC) - mu * mu + LN_EPS);
    const int kt = lane & 1;
    #pragma unroll
    for (int j = 0; j < 32; ++j) {
        float g = (h[j] - mu) * inv * ln_scale[c0 + j] + ln_bias[c0 + j];
        Ast[wave][kt][a_slot(p, j)] = to_bf16(g);
    }
    __syncthreads();

    v16bf a0 = *(const v16bf*)&Ast[wave][0][lane * 16];
    v16bf a1 = *(const v16bf*)&Ast[wave][1][lane * 16];

    v8f acc2[4];
    #pragma unroll
    for (int t = 0; t < 4; ++t) acc2[t] = v8f_zero();

    const int nl = lane & 15, hi = lane >> 4;
    for (int ch = 0; ch < 8; ++ch) {                   // stage-2 K chunks of 32
        #pragma unroll
        for (int sub = 0; sub < 2; ++sub) {
            const int nt = ch * 2 + sub;               // stage-1 output tile
            v16bf b0  = *(const v16bf*)(w1s + ((0 * 16 + nt) * 512 + lane * 16));
            v16bf b1f = *(const v16bf*)(w1s + ((1 * 16 + nt) * 512 + lane * 16));
            v8f g = v8f_zero();
            g = __builtin_amdgcn_wmma_f32_16x16x32_bf16(false, a0, false, b0,  (short)0, g, false, false);
            g = __builtin_amdgcn_wmma_f32_16x16x32_bf16(false, a1, false, b1f, (short)0, g, false, false);
            const float bias = b1[nt * 16 + nl];
            #pragma unroll
            for (int r = 0; r < 8; ++r) {
                int m = r + 8 * hi;
                float v = g[r] + bias;
                v = v > 0.f ? v : 0.f;                 // ReLU
                Gst[wave][a_slot(m, sub * 16 + nl)] = to_bf16(v);
            }
        }
        __syncthreads();
        v16bf ga = *(const v16bf*)&Gst[wave][lane * 16];
        #pragma unroll
        for (int t = 0; t < 4; ++t) {
            v16bf b = *(const v16bf*)(w2s + ((ch * 4 + t) * 512 + lane * 16));
            acc2[t] = __builtin_amdgcn_wmma_f32_16x16x32_bf16(false, ga, false, b, (short)0, acc2[t], false, false);
        }
        __syncthreads();
    }

    // epilogue: + b2, * (layer_scale * LS_VAL), + residual (gather first, then store)
    const float* xin = x   + (size_t)node * (OO * CC);
    float*       op  = out + (size_t)node * (OO * CC);
    float rv[4][8];
    #pragma unroll
    for (int t = 0; t < 4; ++t)
        #pragma unroll
        for (int r = 0; r < 8; ++r)
            rv[t][r] = xin[(r + 8 * hi) * CC + t * 16 + nl];
    #pragma unroll
    for (int t = 0; t < 4; ++t) {
        const int cc = t * 16 + nl;
        const float b2v = b2[cc];
        const float lsv = layer_scale[cc] * LS_VAL;
        #pragma unroll
        for (int r = 0; r < 8; ++r) {
            int m = r + 8 * hi;
            op[m * CC + cc] = (acc2[t][r] + b2v) * lsv + rv[t][r];
        }
    }
}

// ---------------- launch ----------------

extern "C" void kernel_launch(void* const* d_in, const int* in_sizes, int n_in,
                              void* d_out, int out_size, void* d_ws, size_t ws_size,
                              hipStream_t stream) {
    (void)in_sizes; (void)n_in; (void)out_size; (void)ws_size;
    const float* x     = (const float*)d_in[0];
    const float* kb    = (const float*)d_in[1];
    const float* fb    = (const float*)d_in[2];
    const int*   ei    = (const int*)  d_in[3];
    const float* kw    = (const float*)d_in[4];
    const float* fw    = (const float*)d_in[5];
    const float* cbias = (const float*)d_in[6];
    const float* lns   = (const float*)d_in[7];
    const float* lnb   = (const float*)d_in[8];
    const float* w1    = (const float*)d_in[9];
    const float* b1    = (const float*)d_in[10];
    const float* w2    = (const float*)d_in[11];
    const float* b2    = (const float*)d_in[12];
    const float* ls    = (const float*)d_in[13];
    float* out = (float*)d_out;

    char* ws = (char*)d_ws;
    float*          x1  = (float*)(ws);                        // 40,960,000 B
    float*          fk  = (float*)(ws + 40960000);             //     65,536 B
    unsigned short* kws = (unsigned short*)(ws + 41025536);    //      8,192 B
    unsigned short* w1s = (unsigned short*)(ws + 41033728);    //     32,768 B
    unsigned short* w2s = (unsigned short*)(ws + 41066496);    //     32,768 B

    k_zero<<<10000, 256, 0, stream>>>((float4*)x1);            // 10000*256*4 floats exactly
    k_fk  <<<64,    256, 0, stream>>>(fb, fw, fk);             // 16384 outputs
    k_swz <<<16,    256, 0, stream>>>(kw, kws, 64, 64);
    k_swz <<<64,    256, 0, stream>>>(w1, w1s, 64, 256);
    k_swz <<<64,    256, 0, stream>>>(w2, w2s, 256, 64);
    k_edge<<<3125,  256, 0, stream>>>(x, kb, ei, kws, x1);     // 8 waves x 4 edges/block
    k_node<<<1250,  256, 0, stream>>>(x, x1, fk, cbias, lns, lnb,
                                      w1s, b1, w2s, b2, ls, out);
}
